// Qwen3VLVisionModel_28578712387608
// MI455X (gfx1250) — matrix operand order repaired and checked by reading, not compiled, not run on hardware
//
#include <hip/hip_runtime.h>
#include <hip/hip_bf16.h>

typedef __bf16 bf16_t;
typedef __attribute__((ext_vector_type(16))) __bf16 v16bf;
typedef __attribute__((ext_vector_type(8)))  float  v8f;
typedef int v4i_vs __attribute__((vector_size(16)));

#define SEQ     4096
#define HIDDEN  1152
#define INTER   4304
#define HEADS   16
#define HEAD_DIM 72
#define MERGED  4608
#define OUT_H   2048
#define QKV_N   3456
#define ATT_SCALE 0.1178511301977579f   // 72^-0.5

#if defined(__has_builtin)
#if __has_builtin(__builtin_amdgcn_global_load_async_to_lds_b128) && \
    __has_builtin(__builtin_amdgcn_s_wait_asynccnt)
#define USE_ASYNC_LDS 1
#endif
#endif

#define GLOBAL_AS __attribute__((address_space(1)))
#define LDS_AS    __attribute__((address_space(3)))

// ---------- helpers ----------
__device__ inline bf16_t f2bf(float f) { return (bf16_t)f; }

__device__ inline v8f wmma_bf16(v16bf a, v16bf b, v8f c) {
    return __builtin_amdgcn_wmma_f32_16x16x32_bf16(false, a, false, b, (short)0, c, false, false);
}

__device__ inline float gelu_tanh(float x) {
    float x3 = x * x * x;
    return 0.5f * x * (1.0f + tanhf(0.7978845608028654f * (x + 0.044715f * x3)));
}

// block order (32x32 grid, merge 2): i -> (hpos, wpos)
__device__ inline void order_hw(int i, int& hp, int& wp) {
    int d = i & 1, b = (i >> 1) & 1, c = (i >> 2) & 15, a = (i >> 6) & 15;
    hp = a * 2 + b;
    wp = c * 2 + d;
}

#ifdef USE_ASYNC_LDS
__device__ inline void async_b128(const float* gsrc, float* ldst, int imm16) {
    GLOBAL_AS v4i_vs* g = (GLOBAL_AS v4i_vs*)(gsrc);
    LDS_AS    v4i_vs* l = (LDS_AS    v4i_vs*)(ldst);
    if (imm16 == 0)       __builtin_amdgcn_global_load_async_to_lds_b128(g, l, 0,  0);
    else if (imm16 == 16) __builtin_amdgcn_global_load_async_to_lds_b128(g, l, 16, 0);
    else if (imm16 == 32) __builtin_amdgcn_global_load_async_to_lds_b128(g, l, 32, 0);
    else                  __builtin_amdgcn_global_load_async_to_lds_b128(g, l, 48, 0);
}
#endif

// ---------- generic WMMA GEMM:  C[M,N] = act(A[M,K] @ W[N,K]^T + bias) (+resid) ----------
// Double-buffered f32 LDS tiles, async global->LDS staging when available,
// f32->bf16 conversion fused into WMMA fragment assembly.
template<bool GELU, bool RESID>
__global__ __launch_bounds__(256)
void gemm_kernel(const float* __restrict__ A, const float* __restrict__ W,
                 const float* __restrict__ bias, const float* __restrict__ resid,
                 float* __restrict__ C, int M, int N, int K)
{
    __shared__ float As2[2][64][36];    // 64 x 32 f32 (+4 pad)
    __shared__ float Bs2[2][128][36];   // 128 x 32 f32 (+4 pad)

    const int t    = threadIdx.x;
    const int lane = t & 31;
    const int wv   = __builtin_amdgcn_readfirstlane(t >> 5);
    const int wm   = wv >> 2;        // 0..1
    const int wn   = wv & 3;         // 0..3
    const int bm   = blockIdx.y * 64;
    const int bn   = blockIdx.x * 128;

    v8f acc[2][2];
    #pragma unroll
    for (int mi = 0; mi < 2; ++mi)
        #pragma unroll
        for (int ni = 0; ni < 2; ++ni)
            #pragma unroll
            for (int r = 0; r < 8; ++r) acc[mi][ni][r] = 0.0f;

    const int am  = t >> 2;          // 0..63
    const int akc = (t & 3) << 3;    // 0,8,16,24
    const int bnn = t >> 1;          // 0..127
    const int bkc = (t & 1) << 4;    // 0,16

    const int hi8  = (lane & 16) ? 8 : 0;
    const int hi16 = (lane & 16) ? 16 : 0;
    const int KT   = (K + 31) >> 5;

    auto load_tile = [&](int kt, int buf) {
        const int k0 = kt << 5;
        const bool k_full = (k0 + 32 <= K);
        // ---- A tile ----
        {
            const float* ap = A + (size_t)(bm + am) * K + k0 + akc;
            float* dst = &As2[buf][am][akc];
#ifdef USE_ASYNC_LDS
            if (k_full) {
                async_b128(ap, dst, 0);
                async_b128(ap, dst, 16);
            } else
#endif
            {
                #pragma unroll
                for (int j = 0; j < 8; ++j) {
                    int k = k0 + akc + j;
                    dst[j] = (k < K) ? ap[j] : 0.0f;
                }
            }
        }
        // ---- B tile: Bs[n][k] = W[bn+n][k] ----
        {
            int gn = bn + bnn;
            const float* bp = W + (size_t)gn * K + k0 + bkc;
            float* dst = &Bs2[buf][bnn][bkc];
#ifdef USE_ASYNC_LDS
            if (k_full && (bn + 128 <= N)) {
                async_b128(bp, dst, 0);
                async_b128(bp, dst, 16);
                async_b128(bp, dst, 32);
                async_b128(bp, dst, 48);
            } else
#endif
            {
                #pragma unroll
                for (int j = 0; j < 16; ++j) {
                    int k = k0 + bkc + j;
                    dst[j] = (gn < N && k < K) ? bp[j] : 0.0f;
                }
            }
        }
    };

    // prologue
    load_tile(0, 0);
#ifdef USE_ASYNC_LDS
    __builtin_amdgcn_s_wait_asynccnt(0);
#endif
    __syncthreads();

    for (int kt = 0; kt < KT; ++kt) {
        const int cur = kt & 1;
        // prefetch next tile while computing current one
        if (kt + 1 < KT) load_tile(kt + 1, cur ^ 1);

        v16bf afr[2], bfr[2];
        #pragma unroll
        for (int mi = 0; mi < 2; ++mi) {
            const float* p = &As2[cur][wm * 32 + mi * 16 + (lane & 15)][0];
            #pragma unroll
            for (int e = 0; e < 16; ++e) {
                afr[mi][e] = f2bf(p[hi16 + e]);
            }
        }
        #pragma unroll
        for (int ni = 0; ni < 2; ++ni) {
            const float* p = &Bs2[cur][wn * 32 + ni * 16 + (lane & 15)][hi16];
            #pragma unroll
            for (int e = 0; e < 16; ++e) bfr[ni][e] = f2bf(p[e]);
        }
        #pragma unroll
        for (int mi = 0; mi < 2; ++mi)
            #pragma unroll
            for (int ni = 0; ni < 2; ++ni)
                acc[mi][ni] = wmma_bf16(afr[mi], bfr[ni], acc[mi][ni]);

#ifdef USE_ASYNC_LDS
        __builtin_amdgcn_s_wait_asynccnt(0);
#endif
        __syncthreads();
    }

    // epilogue
    #pragma unroll
    for (int mi = 0; mi < 2; ++mi) {
        #pragma unroll
        for (int ni = 0; ni < 2; ++ni) {
            int col = bn + wn * 32 + ni * 16 + (lane & 15);
            if (col < N) {
                float bv = bias[col];
                #pragma unroll
                for (int r = 0; r < 8; ++r) {
                    int row = bm + wm * 32 + mi * 16 + r + hi8;
                    float v = acc[mi][ni][r] + bv;
                    if (GELU) v = gelu_tanh(v);
                    size_t off = (size_t)row * N + col;
                    if (RESID) v += resid[off];
                    C[off] = v;
                }
            }
        }
    }
}

// ---------- LayerNorm: one block per row ----------
__global__ __launch_bounds__(256)
void ln_kernel(const float* __restrict__ x, const float* __restrict__ g,
               const float* __restrict__ b, float* __restrict__ y, int N)
{
    __shared__ float red[256];
    __shared__ float s_mu, s_ri;
    const int t = threadIdx.x;
    const float* xr = x + (size_t)blockIdx.x * N;
    float s = 0.0f;
    for (int i = t; i < N; i += 256) s += xr[i];
    red[t] = s; __syncthreads();
    for (int o = 128; o > 0; o >>= 1) { if (t < o) red[t] += red[t + o]; __syncthreads(); }
    if (t == 0) s_mu = red[0] / (float)N;
    __syncthreads();
    float mu = s_mu;
    s = 0.0f;
    for (int i = t; i < N; i += 256) { float d = xr[i] - mu; s += d * d; }
    red[t] = s; __syncthreads();
    for (int o = 128; o > 0; o >>= 1) { if (t < o) red[t] += red[t + o]; __syncthreads(); }
    if (t == 0) s_ri = rsqrtf(red[0] / (float)N + 1e-6f);
    __syncthreads();
    float ri = s_ri;
    float* yr = y + (size_t)blockIdx.x * N;
    for (int i = t; i < N; i += 256) yr[i] = (xr[i] - mu) * ri * g[i] + b[i];
}

// ---------- patch pos-embed bilinear add ----------
__global__ __launch_bounds__(256)
void posadd_kernel(float* __restrict__ x, const float* __restrict__ pe)
{
    int idx = blockIdx.x * 256 + threadIdx.x;
    if (idx >= SEQ * HIDDEN) return;
    int pos = idx / HIDDEN, ch = idx % HIDDEN;
    int i = pos & 1023, hp, wp;
    order_hw(i, hp, wp);
    float hl = hp * (47.0f / 31.0f), wl = wp * (47.0f / 31.0f);
    int h0 = (int)floorf(hl); int h1 = (h0 + 1 < 47) ? h0 + 1 : 47; float dh = hl - (float)h0;
    int w0 = (int)floorf(wl); int w1 = (w0 + 1 < 47) ? w0 + 1 : 47; float dw = wl - (float)w0;
    float v = (1.0f - dh) * (1.0f - dw) * pe[(size_t)(h0 * 48 + w0) * HIDDEN + ch]
            + (1.0f - dh) * dw          * pe[(size_t)(h0 * 48 + w1) * HIDDEN + ch]
            + dh * (1.0f - dw)          * pe[(size_t)(h1 * 48 + w0) * HIDDEN + ch]
            + dh * dw                   * pe[(size_t)(h1 * 48 + w1) * HIDDEN + ch];
    x[idx] += v;
}

// ---------- RoPE on q,k inside qkv buffer ----------
__global__ __launch_bounds__(256)
void rope_kernel(float* __restrict__ qkv)
{
    int idx = blockIdx.x * 256 + threadIdx.x;   // SEQ * 16 * 36
    if (idx >= SEQ * HEADS * 36) return;
    int pos = idx / (HEADS * 36);
    int rem = idx % (HEADS * 36);
    int hh = rem / 36, tt = rem % 36;
    int i = pos & 1023, hp, wp;
    order_hw(i, hp, wp);
    int   tl  = (tt < 18) ? tt : tt - 18;
    float p   = (float)((tt < 18) ? hp : wp);
    float inv = __expf(-((float)(2 * tl) / 36.0f) * 9.210340371976184f);  // 10000^(-2t/36)
    float ang = p * inv;
    float c = __cosf(ang), s = __sinf(ang);
    size_t base = (size_t)pos * QKV_N + hh * HEAD_DIM + tt;
    float q1 = qkv[base], q2 = qkv[base + 36];
    qkv[base]      = q1 * c - q2 * s;
    qkv[base + 36] = q2 * c + q1 * s;
    size_t kb = base + HIDDEN;
    float k1 = qkv[kb], k2 = qkv[kb + 36];
    qkv[kb]      = k1 * c - k2 * s;
    qkv[kb + 36] = k2 * c + k1 * s;
}

// ---------- attention: one block = (image, head, 32 query rows), full 1024-key softmax in LDS ----------
#define A_QS   0                      // 32*96 bf16      = 6144 B
#define A_KS   6144                   // 64*96 bf16      = 12288 B
#define A_VT   (6144 + 12288)         // 96*72 bf16      = 13824 B
#define A_SC   32256                  // 32*1024 f32     = 131072 B
#define A_PB   (32256 + 131072)       // 32*1032 bf16    = 66048 B
#define A_RED  (163328 + 66048)       // 256 f32         = 1024 B
#define A_RMAX (229376 + 1024)        // 32 f32
#define A_RSUM (230400 + 128)         // 32 f32
#define A_SMEM (230528 + 128)         // total 230656 B

__global__ __launch_bounds__(256)
void attn_kernel(const float* __restrict__ qkv, float* __restrict__ out)
{
    extern __shared__ char smem[];
    bf16_t* Qs  = (bf16_t*)(smem + A_QS);
    bf16_t* Ks  = (bf16_t*)(smem + A_KS);
    bf16_t* Vt  = (bf16_t*)(smem + A_VT);
    float*  Sc  = (float* )(smem + A_SC);
    bf16_t* Pb  = (bf16_t*)(smem + A_PB);
    float*  red = (float* )(smem + A_RED);
    float*  rmx = (float* )(smem + A_RMAX);
    float*  rsm = (float* )(smem + A_RSUM);

    const int t    = threadIdx.x;
    const int lane = t & 31;
    const int wv   = __builtin_amdgcn_readfirstlane(t >> 5);
    const int wm   = wv >> 2;       // 0..1
    const int wn   = wv & 3;        // 0..3
    const int hi8  = (lane & 16) ? 8 : 0;
    const int hi16 = (lane & 16) ? 16 : 0;

    const int qt  = blockIdx.x;     // 0..31
    const int hh  = blockIdx.y;     // 0..15
    const int img = blockIdx.z;     // 0..3
    const int qpos0 = img * 1024 + qt * 32;

    // load Q tile (32 x 96): chunked 8-wide loads, chunks 9..11 are zero-fill
    for (int i = t; i < 32 * 12; i += 256) {
        int r = i / 12, c = i % 12, d0 = c * 8;
        bf16_t* dst = &Qs[r * 96 + d0];
        if (c < 9) {
            const float* src = &qkv[(size_t)(qpos0 + r) * QKV_N + hh * HEAD_DIM + d0];
            #pragma unroll
            for (int j = 0; j < 8; ++j) dst[j] = f2bf(src[j]);
        } else {
            #pragma unroll
            for (int j = 0; j < 8; ++j) dst[j] = f2bf(0.0f);
        }
    }
    __syncthreads();

    // hoist Q fragments (one 16-row tile per wave row group)
    const int qrow = wm * 16 + (lane & 15);
    v16bf qfr[3];
    #pragma unroll
    for (int ks = 0; ks < 3; ++ks) {
        const bf16_t* p = &Qs[qrow * 96 + ks * 32];
        #pragma unroll
        for (int e = 0; e < 16; ++e) { qfr[ks][e] = p[hi16 + e]; }
    }

    // phase 1: scores = Q K^T * scale  -> Sc[32][1024]
    for (int kt = 0; kt < 16; ++kt) {
        for (int i = t; i < 64 * 12; i += 256) {
            int r = i / 12, c = i % 12, d0 = c * 8;
            bf16_t* dst = &Ks[r * 96 + d0];
            if (c < 9) {
                const float* src =
                    &qkv[(size_t)(img * 1024 + kt * 64 + r) * QKV_N + HIDDEN + hh * HEAD_DIM + d0];
                #pragma unroll
                for (int j = 0; j < 8; ++j) dst[j] = f2bf(src[j]);
            } else {
                #pragma unroll
                for (int j = 0; j < 8; ++j) dst[j] = f2bf(0.0f);
            }
        }
        __syncthreads();
        v8f s;
        #pragma unroll
        for (int r = 0; r < 8; ++r) s[r] = 0.0f;
        #pragma unroll
        for (int ks = 0; ks < 3; ++ks) {
            v16bf bfr;
            const bf16_t* p = &Ks[(wn * 16 + (lane & 15)) * 96 + ks * 32 + hi16];
            #pragma unroll
            for (int e = 0; e < 16; ++e) bfr[e] = p[e];
            s = wmma_bf16(qfr[ks], bfr, s);
        }
        int col = kt * 64 + wn * 16 + (lane & 15);
        #pragma unroll
        for (int r = 0; r < 8; ++r)
            Sc[(wm * 16 + r + hi8) * 1024 + col] = s[r] * ATT_SCALE;
        __syncthreads();
    }

    // phase 2: softmax rows (unnormalized exp into Pb bf16, sums kept in LDS)
    {
        int row = t >> 3, seg = t & 7;
        const float* srow = Sc + row * 1024 + seg * 128;
        float m = -3.4e38f;
        for (int j = 0; j < 128; ++j) m = fmaxf(m, srow[j]);
        red[row * 8 + seg] = m; __syncthreads();
        if (seg == 0) {
            float mm = red[row * 8];
            for (int j = 1; j < 8; ++j) mm = fmaxf(mm, red[row * 8 + j]);
            rmx[row] = mm;
        }
        __syncthreads();
        float mm = rmx[row], ssum = 0.0f;
        bf16_t* prow = Pb + row * 1032 + seg * 128;
        for (int j = 0; j < 128; ++j) {
            float e = __expf(srow[j] - mm);
            prow[j] = f2bf(e);
            ssum += e;
        }
        red[row * 8 + seg] = ssum; __syncthreads();
        if (seg == 0) {
            float sm = 0.0f;
            for (int j = 0; j < 8; ++j) sm += red[row * 8 + j];
            rsm[row] = sm;
        }
        __syncthreads();
    }

    // phase 3: O = P V   (V^T tile in LDS: [96 dims][64 keys])
    v8f acc0, acc1;
    #pragma unroll
    for (int r = 0; r < 8; ++r) { acc0[r] = 0.0f; acc1[r] = 0.0f; }
    for (int kt = 0; kt < 16; ++kt) {
        // chunked 8-wide loads of V rows, transposed scatter into Vt
        for (int i = t; i < 64 * 12; i += 256) {
            int k = i / 12, c = i % 12, d0 = c * 8;
            if (c < 9) {
                const float* src =
                    &qkv[(size_t)(img * 1024 + kt * 64 + k) * QKV_N + 2 * HIDDEN + hh * HEAD_DIM + d0];
                #pragma unroll
                for (int j = 0; j < 8; ++j) Vt[(d0 + j) * 72 + k] = f2bf(src[j]);
            } else {
                #pragma unroll
                for (int j = 0; j < 8; ++j) Vt[(d0 + j) * 72 + k] = f2bf(0.0f);
            }
        }
        __syncthreads();
        #pragma unroll
        for (int ks = 0; ks < 2; ++ks) {
            v16bf pfr;
            const bf16_t* pp = &Pb[qrow * 1032 + kt * 64 + ks * 32];
            #pragma unroll
            for (int e = 0; e < 16; ++e) { pfr[e] = pp[hi16 + e]; }
            v16bf vfr;
            const bf16_t* vp = &Vt[(wn * 16 + (lane & 15)) * 72 + ks * 32 + hi16];
            #pragma unroll
            for (int e = 0; e < 16; ++e) vfr[e] = vp[e];
            acc0 = wmma_bf16(pfr, vfr, acc0);
            if (wn == 0) {            // extra column tile: dims 64..79 (covers 64..71)
                v16bf vfr2;
                const bf16_t* vp2 = &Vt[(64 + (lane & 15)) * 72 + ks * 32 + hi16];
                #pragma unroll
                for (int e = 0; e < 16; ++e) vfr2[e] = vp2[e];
                acc1 = wmma_bf16(pfr, vfr2, acc1);
            }
        }
        __syncthreads();
    }

    // epilogue: normalize by row sum, write out[pos][h*72+d]
    #pragma unroll
    for (int r = 0; r < 8; ++r) {
        int rr = wm * 16 + r + hi8;
        float rinv = 1.0f / rsm[rr];
        size_t pos = (size_t)(qpos0 + rr);
        int d0 = wn * 16 + (lane & 15);
        if (d0 < HEAD_DIM) out[pos * HIDDEN + hh * HEAD_DIM + d0] = acc0[r] * rinv;
        if (wn == 0) {
            int d1 = 64 + (lane & 15);
            if (d1 < HEAD_DIM) out[pos * HIDDEN + hh * HEAD_DIM + d1] = acc1[r] * rinv;
        }
    }
}

// ---------- host orchestration ----------
static void launch_gemm(hipStream_t s, const float* A, const float* W, const float* bias,
                        const float* resid, float* C, int M, int N, int K,
                        bool do_gelu, bool do_resid)
{
    dim3 grid((N + 127) / 128, M / 64);
    if (do_gelu)       gemm_kernel<true,  false><<<grid, 256, 0, s>>>(A, W, bias, nullptr, C, M, N, K);
    else if (do_resid) gemm_kernel<false, true ><<<grid, 256, 0, s>>>(A, W, bias, resid,   C, M, N, K);
    else               gemm_kernel<false, false><<<grid, 256, 0, s>>>(A, W, bias, nullptr, C, M, N, K);
}

extern "C" void kernel_launch(void* const* d_in, const int* in_sizes, int n_in,
                              void* d_out, int out_size, void* d_ws, size_t ws_size,
                              hipStream_t stream)
{
    const float* pixel   = (const float*)d_in[0];
    const float* patch_w = (const float*)d_in[1];
    const float* patch_b = (const float*)d_in[2];
    const float* pos_emb = (const float*)d_in[3];
    const float* qkv_w   = (const float*)d_in[4];
    const float* qkv_b   = (const float*)d_in[5];
    const float* proj_w  = (const float*)d_in[6];
    const float* proj_b  = (const float*)d_in[7];
    const float* ln1_g   = (const float*)d_in[8];
    const float* ln1_b   = (const float*)d_in[9];
    const float* ln2_g   = (const float*)d_in[10];
    const float* ln2_b   = (const float*)d_in[11];
    const float* fc1_w   = (const float*)d_in[12];
    const float* fc1_b   = (const float*)d_in[13];
    const float* fc2_w   = (const float*)d_in[14];
    const float* fc2_b   = (const float*)d_in[15];
    const float* m_ln_g  = (const float*)d_in[16];
    const float* m_ln_b  = (const float*)d_in[17];
    const float* m_fc1_w = (const float*)d_in[18];
    const float* m_fc1_b = (const float*)d_in[19];
    const float* m_fc2_w = (const float*)d_in[20];
    const float* m_fc2_b = (const float*)d_in[21];
    const float* ds_ln_g = (const float*)d_in[22];
    const float* ds_ln_b = (const float*)d_in[23];
    const float* ds_fc1w = (const float*)d_in[24];
    const float* ds_fc1b = (const float*)d_in[25];
    const float* ds_fc2w = (const float*)d_in[26];
    const float* ds_fc2b = (const float*)d_in[27];
    float* out = (float*)d_out;

    // workspace layout (floats)
    float* x    = (float*)d_ws;                          // 4096*1152
    float* xn   = x    + (size_t)SEQ * HIDDEN;           // 4096*1152 (also 1024*4608)
    float* qkv  = xn   + (size_t)SEQ * HIDDEN;           // 4096*3456
    float* atto = qkv  + (size_t)SEQ * QKV_N;            // 4096*1152
    float* h    = atto + (size_t)SEQ * HIDDEN;           // 4096*4304 (also 1024*4608)

    // patch embed + positional embedding
    launch_gemm(stream, pixel, patch_w, patch_b, nullptr, x, SEQ, HIDDEN, 1536, false, false);
    posadd_kernel<<<(SEQ * HIDDEN + 255) / 256, 256, 0, stream>>>(x, pos_emb);

    for (int i = 0; i < 6; ++i) {
        // attention block
        ln_kernel<<<SEQ, 256, 0, stream>>>(x, ln1_g + i * HIDDEN, ln1_b + i * HIDDEN, xn, HIDDEN);
        launch_gemm(stream, xn, qkv_w + (size_t)i * QKV_N * HIDDEN, qkv_b + (size_t)i * QKV_N,
                    nullptr, qkv, SEQ, QKV_N, HIDDEN, false, false);
        rope_kernel<<<(SEQ * HEADS * 36 + 255) / 256, 256, 0, stream>>>(qkv);
        attn_kernel<<<dim3(32, 16, 4), 256, A_SMEM, stream>>>(qkv, atto);
        launch_gemm(stream, atto, proj_w + (size_t)i * HIDDEN * HIDDEN, proj_b + (size_t)i * HIDDEN,
                    x, x, SEQ, HIDDEN, HIDDEN, false, true);
        // MLP block
        ln_kernel<<<SEQ, 256, 0, stream>>>(x, ln2_g + i * HIDDEN, ln2_b + i * HIDDEN, xn, HIDDEN);
        launch_gemm(stream, xn, fc1_w + (size_t)i * INTER * HIDDEN, fc1_b + (size_t)i * INTER,
                    nullptr, h, SEQ, INTER, HIDDEN, true, false);
        launch_gemm(stream, h, fc2_w + (size_t)i * HIDDEN * INTER, fc2_b + (size_t)i * HIDDEN,
                    x, x, SEQ, HIDDEN, INTER, false, true);
        // deepstack mergers (post-LN over merged rows)
        if (i == 2 || i == 4) {
            int j = (i == 2) ? 0 : 1;
            ln_kernel<<<1024, 256, 0, stream>>>(x, ds_ln_g + (size_t)j * MERGED,
                                                ds_ln_b + (size_t)j * MERGED, xn, MERGED);
            launch_gemm(stream, xn, ds_fc1w + (size_t)j * MERGED * MERGED,
                        ds_fc1b + (size_t)j * MERGED, nullptr, h, 1024, MERGED, MERGED, true, false);
            launch_gemm(stream, h, ds_fc2w + (size_t)j * OUT_H * MERGED,
                        ds_fc2b + (size_t)j * OUT_H, nullptr,
                        out + (size_t)(1 + j) * 1024 * OUT_H, 1024, OUT_H, MERGED, false, false);
        }
    }

    // final merger: LN per 1152-row, then merged MLP
    ln_kernel<<<SEQ, 256, 0, stream>>>(x, m_ln_g, m_ln_b, xn, HIDDEN);
    launch_gemm(stream, xn, m_fc1_w, m_fc1_b, nullptr, h, 1024, MERGED, MERGED, true, false);
    launch_gemm(stream, h, m_fc2_w, m_fc2_b, nullptr, out, 1024, OUT_H, MERGED, false, false);
}
